// HolCCG_45226005627037
// MI455X (gfx1250) — compile-verified, matrix operand order checked by reading, not run on hardware
//
#include <hip/hip_runtime.h>
#include <hip/hip_bf16.h>
#include <math.h>

// ---------------------------------------------------------------------------
// Problem constants (match reference)
// ---------------------------------------------------------------------------
#define BB   128
#define LL   128
#define NNOD 255
#define DD   1024
#define KK   32
#define NWRD 512
#define NPHR 512
#define NORMV 30.0f

typedef __bf16 bf16;
typedef __attribute__((ext_vector_type(16))) __bf16 v16bf;
typedef __attribute__((ext_vector_type(8)))  __bf16 v8bf;
typedef __attribute__((ext_vector_type(8)))  float  v8f;
typedef __attribute__((ext_vector_type(4)))  int    v4i;

union FragBF { v16bf v; v8bf h[2]; };

__device__ inline v8bf zero_v8bf() {
  union { unsigned int u[4]; v8bf v; } z;
  z.u[0] = z.u[1] = z.u[2] = z.u[3] = 0u;
  return z.v;
}

// ---------------------------------------------------------------------------
// CDNA5 async global->LDS copy (GLOBAL_LOAD_ASYNC_TO_LDS_B128, ASYNCcnt).
// Builtin signature (from hipcc diagnostic): (v4i as1*, v4i as3*, Ii, Ii).
// Falls back to a synchronous VGPR-staged copy if the builtin is absent.
// Pointers are materialized via integer casts: flat LDS addresses carry the
// LDS byte offset in bits [31:0] (ISA 10.2 aperture rules); global flat
// addresses are identity-mapped.
// ---------------------------------------------------------------------------
#if defined(__AMDGCN__) && __has_builtin(__builtin_amdgcn_global_load_async_to_lds_b128)
#define USE_ASYNC 1
#else
#define USE_ASYNC 0
#endif

#if USE_ASYNC
#define AS_G __attribute__((address_space(1)))
#define AS_L __attribute__((address_space(3)))
__device__ inline void async_copy16(const bf16* g, bf16* l) {
  __builtin_amdgcn_global_load_async_to_lds_b128(
      (AS_G v4i*)(unsigned long long)(const void*)g,
      (AS_L v4i*)(unsigned int)(unsigned long long)(void*)l,
      0, 0);
}
#else
__device__ inline void async_copy16(const bf16* g, bf16* l) {
  *(v8bf*)l = *(const v8bf*)g;
}
#endif

template <int N> __device__ inline void wait_async() {
#if USE_ASYNC
#if __has_builtin(__builtin_amdgcn_s_wait_asynccnt)
  __builtin_amdgcn_s_wait_asynccnt(N);
#else
  asm volatile("s_wait_asynccnt %0" :: "n"(N) : "memory");
#endif
#endif
}

// ---------------------------------------------------------------------------
// f32 -> bf16 conversion (vectorized, grid-stride)
// ---------------------------------------------------------------------------
__global__ void k_f32_to_bf16(const float* __restrict__ in, bf16* __restrict__ out,
                              long long n4) {
  long long i = (long long)blockIdx.x * blockDim.x + threadIdx.x;
  long long stride = (long long)gridDim.x * blockDim.x;
  for (; i < n4; i += stride) {
    float4 v = ((const float4*)in)[i];
    bf16* o = out + i * 4;
    o[0] = (bf16)v.x; o[1] = (bf16)v.y; o[2] = (bf16)v.z; o[3] = (bf16)v.w;
  }
}

// ---------------------------------------------------------------------------
// Generic GEMM:  C[M x N] = A[M x K] * W[N x K]^T (+ bias[N])
// A, W in bf16 row-major; C f32 row-major (ldc == N).
// Block: 256 threads = 8 waves; tile 128(M) x 128(N); wave does 32x64.
// v_wmma_f32_16x16x32_bf16 with f32 accumulation; global->LDS staging via
// CDNA5 async-to-LDS copies, double buffered so tile t+1 DMA overlaps the
// 8 WMMAs of tile t.
// Requires: M % 128 == 0, K % 32 == 0. N arbitrary (guarded).
// ---------------------------------------------------------------------------
#define LDS_STR 40   // padded bf16 row stride (80B) -> conflict-free b128 reads

__global__ void k_gemm_bf16(const bf16* __restrict__ A, const bf16* __restrict__ W,
                            const float* __restrict__ bias, float* __restrict__ C,
                            int M, int N, int Kd) {
  __shared__ bf16 As[2][128 * LDS_STR];
  __shared__ bf16 Bs[2][128 * LDS_STR];

  const int tid   = threadIdx.x;
  const int wave  = tid >> 5;
  const int lane  = tid & 31;
  const int waveM = wave >> 1;        // 0..3  (rows of 32)
  const int waveN = wave & 1;         // 0..1  (cols of 64)
  const int half  = lane >> 4;        // K-half select per ISA bf16 A layout
  const int r     = lane & 15;

  const int m0 = blockIdx.y * 128;
  const int n0 = blockIdx.x * 128;

  v8f acc[2][4];
  for (int i = 0; i < 2; ++i)
    for (int j = 0; j < 4; ++j)
      for (int e = 0; e < 8; ++e) acc[i][j][e] = 0.0f;

  const int lr = tid >> 2;            // loader row (0..63), +64 for second half
  const int cg = (tid & 3) * 8;       // loader col group (8 bf16 = 16B)

  // Issue the 4 per-thread async copies for one 128x32 A/B tile pair.
  auto issue_tile = [&](int kc, int p) {
    for (int rr = lr; rr < 128; rr += 64)
      async_copy16(A + (long long)(m0 + rr) * Kd + kc + cg,
                   &As[p][rr * LDS_STR + cg]);
    for (int rr = lr; rr < 128; rr += 64) {
      if (n0 + rr < N)
        async_copy16(W + (long long)(n0 + rr) * Kd + kc + cg,
                     &Bs[p][rr * LDS_STR + cg]);
      else
        *(v8bf*)&Bs[p][rr * LDS_STR + cg] = zero_v8bf();
    }
  };

  const int nk = Kd >> 5;             // K tiles of 32
  issue_tile(0, 0);

  for (int t = 0; t < nk; ++t) {
    const int p = t & 1;
    if (t + 1 < nk) {
      issue_tile((t + 1) << 5, p ^ 1);   // prefetch next tile (overlaps WMMA)
      wait_async<4>();                   // tile t's copies are now complete
    } else {
      wait_async<0>();
    }
    __syncthreads();

    FragBF aF[2], bF[4];
    for (int i = 0; i < 2; ++i) {
      int row = waveM * 32 + i * 16 + r;
      aF[i].h[0] = *(const v8bf*)&As[p][row * LDS_STR + half * 8];
      aF[i].h[1] = *(const v8bf*)&As[p][row * LDS_STR + 16 + half * 8];
    }
    for (int j = 0; j < 4; ++j) {
      int row = waveN * 64 + j * 16 + r;
      bF[j].h[0] = *(const v8bf*)&Bs[p][row * LDS_STR + half * 8];
      bF[j].h[1] = *(const v8bf*)&Bs[p][row * LDS_STR + 16 + half * 8];
    }
    for (int i = 0; i < 2; ++i)
      for (int j = 0; j < 4; ++j)
        acc[i][j] = __builtin_amdgcn_wmma_f32_16x16x32_bf16(
            false, aF[i].v, false, bF[j].v, (short)0, acc[i][j], false, false);

    __syncthreads();                     // reads done before next prefetch lands
  }

  // Epilogue. C/D layout: lane<16 -> (m=vgpr, n=lane); lane>=16 -> (m=8+vgpr, n=lane-16)
  const int nSub  = lane & 15;
  const int mBase = (lane >> 4) * 8;
  for (int i = 0; i < 2; ++i)
    for (int j = 0; j < 4; ++j) {
      int gn = n0 + waveN * 64 + j * 16 + nSub;
      if (gn >= N) continue;
      float bv = bias ? bias[gn] : 0.0f;
      int gm0 = m0 + waveM * 32 + i * 16 + mBase;
      for (int e = 0; e < 8; ++e) {
        int gm = gm0 + e;
        if (gm < M) C[(long long)gm * N + gn] = acc[i][j][e] + bv;
      }
    }
}

// ---------------------------------------------------------------------------
// Row-wise L2 norm: out = 30*x/(||x||+1e-12). Writes f32 and bf16 copies.
// One 256-thread block per row of 1024.
// ---------------------------------------------------------------------------
__global__ void k_l2norm_row(const float* __restrict__ in, float* __restrict__ outF,
                             bf16* __restrict__ outB) {
  __shared__ float red[256];
  const long long row = blockIdx.x;
  const int tid = threadIdx.x;
  float4 v = ((const float4*)(in + row * DD))[tid];
  float s = v.x * v.x + v.y * v.y + v.z * v.z + v.w * v.w;
  red[tid] = s; __syncthreads();
  for (int off = 128; off > 0; off >>= 1) {
    if (tid < off) red[tid] += red[tid + off];
    __syncthreads();
  }
  float scale = NORMV / (sqrtf(red[0]) + 1e-12f);
  v.x *= scale; v.y *= scale; v.z *= scale; v.w *= scale;
  ((float4*)(outF + row * DD))[tid] = v;
  bf16* ob = outB + row * DD + tid * 4;
  ob[0] = (bf16)v.x; ob[1] = (bf16)v.y; ob[2] = (bf16)v.z; ob[3] = (bf16)v.w;
}

// ---------------------------------------------------------------------------
// Row-wise LayerNorm + ReLU -> bf16. One 256-thread block per row of 1024.
// ---------------------------------------------------------------------------
__global__ void k_layernorm_relu(const float* __restrict__ in,
                                 const float* __restrict__ g,
                                 const float* __restrict__ be,
                                 bf16* __restrict__ outB) {
  __shared__ float redS[256];
  __shared__ float redQ[256];
  const long long row = blockIdx.x;
  const int tid = threadIdx.x;
  float4 v = ((const float4*)(in + row * DD))[tid];
  redS[tid] = v.x + v.y + v.z + v.w;
  redQ[tid] = v.x * v.x + v.y * v.y + v.z * v.z + v.w * v.w;
  __syncthreads();
  for (int off = 128; off > 0; off >>= 1) {
    if (tid < off) { redS[tid] += redS[tid + off]; redQ[tid] += redQ[tid + off]; }
    __syncthreads();
  }
  float mu  = redS[0] * (1.0f / DD);
  float var = redQ[0] * (1.0f / DD) - mu * mu;
  float rs  = rsqrtf(var + 1e-5f);
  float4 gg = ((const float4*)g)[tid];
  float4 bb = ((const float4*)be)[tid];
  float h0 = fmaxf((v.x - mu) * rs * gg.x + bb.x, 0.0f);
  float h1 = fmaxf((v.y - mu) * rs * gg.y + bb.y, 0.0f);
  float h2 = fmaxf((v.z - mu) * rs * gg.z + bb.z, 0.0f);
  float h3 = fmaxf((v.w - mu) * rs * gg.w + bb.w, 0.0f);
  bf16* ob = outB + row * DD + tid * 4;
  ob[0] = (bf16)h0; ob[1] = (bf16)h1; ob[2] = (bf16)h2; ob[3] = (bf16)h3;
}

// ---------------------------------------------------------------------------
// Compose chain: 128 persistent blocks (one per batch).
// Chain (from reference construction): node L+i = l2norm(circ_corr(prev, leaf[i+1]))
// with prev = leaf[0] for i==0. circ_corr(a,b)[k] = sum_j a[j]*b[(j+k)%D].
// Running vector kept in LDS (f32); phrase rows emitted bf16 into span A buffer
// at row b*(L-1)+i. rvec == vec in the reference (identical composition info),
// so one pass suffices.
// ---------------------------------------------------------------------------
__global__ void k_compose(const float* __restrict__ wv, bf16* __restrict__ spanA) {
  __shared__ float a[DD];
  __shared__ float bdup[2 * DD];
  __shared__ float red[256];
  const int b = blockIdx.x;
  const int tid = threadIdx.x;
  const int k0 = tid, k1 = tid + 256, k2 = tid + 512, k3 = tid + 768;

  for (int e = tid; e < DD; e += 256)
    a[e] = wv[((long long)b * LL + 0) * DD + e];

  for (int i = 0; i < LL - 1; ++i) {
    __syncthreads();
    const float* leaf = wv + ((long long)b * LL + (i + 1)) * DD;
    for (int e = tid; e < DD; e += 256) {
      float v = leaf[e];
      bdup[e] = v; bdup[e + DD] = v;
    }
    __syncthreads();

    float c0 = 0.f, c1 = 0.f, c2 = 0.f, c3 = 0.f;
    for (int j = 0; j < DD; j += 4) {
      float4 av = *(const float4*)&a[j];
      c0 += av.x * bdup[j + 0 + k0]; c1 += av.x * bdup[j + 0 + k1];
      c2 += av.x * bdup[j + 0 + k2]; c3 += av.x * bdup[j + 0 + k3];
      c0 += av.y * bdup[j + 1 + k0]; c1 += av.y * bdup[j + 1 + k1];
      c2 += av.y * bdup[j + 1 + k2]; c3 += av.y * bdup[j + 1 + k3];
      c0 += av.z * bdup[j + 2 + k0]; c1 += av.z * bdup[j + 2 + k1];
      c2 += av.z * bdup[j + 2 + k2]; c3 += av.z * bdup[j + 2 + k3];
      c0 += av.w * bdup[j + 3 + k0]; c1 += av.w * bdup[j + 3 + k1];
      c2 += av.w * bdup[j + 3 + k2]; c3 += av.w * bdup[j + 3 + k3];
    }

    red[tid] = c0 * c0 + c1 * c1 + c2 * c2 + c3 * c3;
    __syncthreads();
    for (int off = 128; off > 0; off >>= 1) {
      if (tid < off) red[tid] += red[tid + off];
      __syncthreads();
    }
    float scale = NORMV / (sqrtf(red[0]) + 1e-12f);
    float s0 = c0 * scale, s1 = c1 * scale, s2 = c2 * scale, s3 = c3 * scale;
    a[k0] = s0; a[k1] = s1; a[k2] = s2; a[k3] = s3;
    bf16* dst = spanA + ((long long)b * (LL - 1) + i) * DD;
    dst[k0] = (bf16)s0; dst[k1] = (bf16)s1; dst[k2] = (bf16)s2; dst[k3] = (bf16)s3;
  }
}

// ---------------------------------------------------------------------------
// Gather negatives: neg row idx = b*K+k copies phrase row (id-L) of batch b
// (rvec == vec) into spanA rows [B*(L-1), B*(L-1)+B*K). One block per row.
// ---------------------------------------------------------------------------
__global__ void k_gather_neg(const int* __restrict__ ids, bf16* __restrict__ spanA) {
  const int idx = blockIdx.x;                    // 0 .. B*K-1
  const int b = idx / KK, k = idx % KK;
  int id = ids[b * KK + k];
  if (id < LL) id = LL;
  if (id >= NNOD) id = NNOD - 1;
  const long long srow = (long long)b * (LL - 1) + (id - LL);
  const long long drow = (long long)BB * (LL - 1) + idx;
  const uint2* s2 = (const uint2*)(spanA + srow * DD);
  uint2* d2 = (uint2*)(spanA + drow * DD);
  d2[threadIdx.x] = s2[threadIdx.x];             // 256 * 8B = 2048B = 1024 bf16
}

// ---------------------------------------------------------------------------
// Labels -> float into tail of d_out.
// ---------------------------------------------------------------------------
__global__ void k_labels(const int* __restrict__ wl, const int* __restrict__ pl,
                         float* __restrict__ outW, float* __restrict__ outP,
                         float* __restrict__ outS) {
  const int nW = BB * LL, nP = BB * (LL - 1), nS = BB * (LL - 1) + BB * KK;
  long long i = (long long)blockIdx.x * blockDim.x + threadIdx.x;
  long long stride = (long long)gridDim.x * blockDim.x;
  for (; i < nW + nP + nS; i += stride) {
    if (i < nW)            outW[i] = (float)wl[i];
    else if (i < nW + nP)  outP[i - nW] = (float)pl[i - nW];
    else {
      long long j = i - nW - nP;
      outS[j] = (j < nP) ? 1.0f : 0.0f;
    }
  }
}

// ---------------------------------------------------------------------------
// Host-side orchestration
// ---------------------------------------------------------------------------
extern "C" void kernel_launch(void* const* d_in, const int* in_sizes, int n_in,
                              void* d_out, int out_size, void* d_ws, size_t ws_size,
                              hipStream_t stream) {
  (void)in_sizes; (void)n_in; (void)out_size; (void)ws_size;

  const float* word_vector = (const float*)d_in[0];
  const int*   neg_id      = (const int*)d_in[4];
  const int*   word_label  = (const int*)d_in[5];
  const int*   phrase_lbl  = (const int*)d_in[6];
  const float* enc_W = (const float*)d_in[7];
  const float* enc_b = (const float*)d_in[8];
  const float* wc_W1 = (const float*)d_in[9];
  const float* wc_b1 = (const float*)d_in[10];
  const float* wc_g  = (const float*)d_in[11];
  const float* wc_be = (const float*)d_in[12];
  const float* wc_W2 = (const float*)d_in[13];
  const float* wc_b2 = (const float*)d_in[14];
  const float* pc_W1 = (const float*)d_in[15];
  const float* pc_b1 = (const float*)d_in[16];
  const float* pc_g  = (const float*)d_in[17];
  const float* pc_be = (const float*)d_in[18];
  const float* pc_W2 = (const float*)d_in[19];
  const float* pc_b2 = (const float*)d_in[20];
  const float* sc_W1 = (const float*)d_in[21];
  const float* sc_b1 = (const float*)d_in[22];
  const float* sc_g  = (const float*)d_in[23];
  const float* sc_be = (const float*)d_in[24];
  const float* sc_W2 = (const float*)d_in[25];
  const float* sc_b2 = (const float*)d_in[26];

  const long long M_word = (long long)BB * LL;                   // 16384
  const long long M_phr  = (long long)BB * (LL - 1);             // 16256
  const long long M_span = M_phr + (long long)BB * KK;           // 20352

  // ---- workspace layout (bytes), ~311 MB total ----
  char* ws = (char*)d_ws;
  size_t off = 0;
  auto alignup = [](size_t x) { return (x + 255) & ~(size_t)255; };
  #define WS_ALLOC(ptrtype, name, bytes) \
    ptrtype name = (ptrtype)(ws + off); off = alignup(off + (bytes));
  WS_ALLOC(bf16*, wvin_b,  M_word * DD * 2)            // word_vector bf16
  WS_ALLOC(bf16*, encW_b,  (size_t)DD * DD * 2)
  WS_ALLOC(bf16*, wcW1_b,  (size_t)DD * DD * 2)
  WS_ALLOC(bf16*, pcW1_b,  (size_t)DD * DD * 2)
  WS_ALLOC(bf16*, scW1_b,  (size_t)DD * DD * 2)
  WS_ALLOC(bf16*, wcW2_b,  (size_t)NWRD * DD * 2)
  WS_ALLOC(bf16*, pcW2_b,  (size_t)NPHR * DD * 2)
  WS_ALLOC(bf16*, scW2_b,  (size_t)2 * DD * 2)
  WS_ALLOC(float*, wv_f,   M_word * DD * 4)            // l2norm'd encoder out
  WS_ALLOC(bf16*,  wv_b,   M_word * DD * 2)
  WS_ALLOC(bf16*,  spanA,  M_span * DD * 2)            // phrase rows + neg rows
  WS_ALLOC(float*, tmp_f,  M_span * DD * 4)            // GEMM scratch (f32)
  WS_ALLOC(bf16*,  h_b,    M_span * DD * 2)            // post-LN hidden (bf16)
  #undef WS_ALLOC

  float* out = (float*)d_out;
  float* word_out   = out;                              // 16384*512
  float* phrase_out = out + M_word * NWRD;              // 16256*512
  float* span_out   = phrase_out + M_phr * NPHR;        // 20352*2
  float* lblW = span_out + M_span * 2;
  float* lblP = lblW + M_word;
  float* lblS = lblP + M_phr;

  const dim3 blk(256);
  auto cvt = [&](const float* src, bf16* dst, long long n) {
    long long n4 = n / 4;
    int g = (int)((n4 + 255) / 256);
    k_f32_to_bf16<<<dim3(g), blk, 0, stream>>>(src, dst, n4);
  };

  // 1) convert inputs/weights to bf16
  cvt(word_vector, wvin_b, M_word * DD);
  cvt(enc_W, encW_b, (long long)DD * DD);
  cvt(wc_W1, wcW1_b, (long long)DD * DD);
  cvt(pc_W1, pcW1_b, (long long)DD * DD);
  cvt(sc_W1, scW1_b, (long long)DD * DD);
  cvt(wc_W2, wcW2_b, (long long)NWRD * DD);
  cvt(pc_W2, pcW2_b, (long long)NPHR * DD);
  cvt(sc_W2, scW2_b, (long long)2 * DD);

  // 2) encoder GEMM + l2norm -> wv (f32 + bf16)
  k_gemm_bf16<<<dim3(DD / 128, (int)(M_word / 128)), blk, 0, stream>>>(
      wvin_b, encW_b, enc_b, tmp_f, (int)M_word, DD, DD);
  k_l2norm_row<<<dim3((int)M_word), blk, 0, stream>>>(tmp_f, wv_f, wv_b);

  // 3) compose chain -> phrase rows (bf16) in spanA[0 : 16256)
  k_compose<<<dim3(BB), blk, 0, stream>>>(wv_f, spanA);

  // 4) gather negatives (rvec == vec) -> spanA[16256 : 20352)
  k_gather_neg<<<dim3(BB * KK), blk, 0, stream>>>(neg_id, spanA);

  // 5) word classifier
  k_gemm_bf16<<<dim3(DD / 128, (int)(M_word / 128)), blk, 0, stream>>>(
      wv_b, wcW1_b, wc_b1, tmp_f, (int)M_word, DD, DD);
  k_layernorm_relu<<<dim3((int)M_word), blk, 0, stream>>>(tmp_f, wc_g, wc_be, h_b);
  k_gemm_bf16<<<dim3(NWRD / 128, (int)(M_word / 128)), blk, 0, stream>>>(
      h_b, wcW2_b, wc_b2, word_out, (int)M_word, NWRD, DD);

  // 6) phrase classifier (first 16256 rows of spanA)
  k_gemm_bf16<<<dim3(DD / 128, (int)(M_phr / 128)), blk, 0, stream>>>(
      spanA, pcW1_b, pc_b1, tmp_f, (int)M_phr, DD, DD);
  k_layernorm_relu<<<dim3((int)M_phr), blk, 0, stream>>>(tmp_f, pc_g, pc_be, h_b);
  k_gemm_bf16<<<dim3(NPHR / 128, (int)(M_phr / 128)), blk, 0, stream>>>(
      h_b, pcW2_b, pc_b2, phrase_out, (int)M_phr, NPHR, DD);

  // 7) span classifier (all 20352 rows of spanA), N=2 head (guarded tiles)
  k_gemm_bf16<<<dim3(DD / 128, (int)(M_span / 128)), blk, 0, stream>>>(
      spanA, scW1_b, sc_b1, tmp_f, (int)M_span, DD, DD);
  k_layernorm_relu<<<dim3((int)M_span), blk, 0, stream>>>(tmp_f, sc_g, sc_be, h_b);
  k_gemm_bf16<<<dim3(1, (int)(M_span / 128)), blk, 0, stream>>>(
      h_b, scW2_b, sc_b2, span_out, (int)M_span, 2, DD);

  // 8) labels
  {
    long long tot = M_word + M_phr + M_span;
    int g = (int)((tot + 255) / 256);
    k_labels<<<dim3(g), blk, 0, stream>>>(word_label, phrase_lbl, lblW, lblP, lblS);
  }
}
